// Qwen3MoeAttention_61710090109455
// MI455X (gfx1250) — matrix operand order repaired and checked by reading, not compiled, not run on hardware
//
#include <hip/hip_runtime.h>

// ---------------------------------------------------------------------------
// Qwen3-MoE fused QKV projection + per-head RMSNorm + RoPE for gfx1250.
// Pass 1 (if ws fits): f32 -> bf16 pre-convert of activations + weights.
// Pass 2: bf16 WMMA GEMM (f32 accum), per-head RMSNorm + RoPE epilogue.
// Block = 128 rows x 128 cols (one head strip). Wave tile = 32x64 (8 accs).
// Staging: double-buffered GLOBAL_LOAD_ASYNC_TO_LDS_B128 pipeline (ASYNCcnt),
// so each chunk's global->LDS transfer overlaps the previous chunk's WMMAs.
// ---------------------------------------------------------------------------

typedef __bf16 bf16_t;
typedef __bf16 v16bf __attribute__((ext_vector_type(16)));
typedef __bf16 v8bf  __attribute__((ext_vector_type(8)));
typedef __bf16 v4bf  __attribute__((ext_vector_type(4)));
typedef float  v8f   __attribute__((ext_vector_type(8)));
typedef float  v4f   __attribute__((ext_vector_type(4)));
typedef int    v4i   __attribute__((ext_vector_type(4)));

#define AS1 __attribute__((address_space(1)))
#define AS3 __attribute__((address_space(3)))

#define T_ROWS 16384
#define HIDDEN 4096
#define NTOT   5120
#define HEAD   128
#define MT     128        // rows per block
#define KC     64         // K chunk (2 WMMA k-steps)
#define NCHUNK (HIDDEN / KC)
#define LDA    72         // LDS pitch in bf16 (KC + 8, keeps 16B alignment)
#define LDC    132        // LDS pitch for f32 C half-tile
#define BUFE   (2 * 128 * LDA)   // bf16 elems per (A+B) buffer = 18432

#if defined(__gfx1250__) && __has_builtin(__builtin_amdgcn_global_load_async_to_lds_b128)
#define HAVE_ASYNC_LDS 1
#else
#define HAVE_ASYNC_LDS 0
#endif

__device__ __forceinline__ void wait_async0() {
#if HAVE_ASYNC_LDS
#if __has_builtin(__builtin_amdgcn_s_wait_asynccnt)
    __builtin_amdgcn_s_wait_asynccnt(0);
#else
    asm volatile("s_wait_asynccnt 0x0" ::: "memory");
#endif
#endif
}

// 16-byte global -> LDS copy; async (ASYNCcnt) when available.
__device__ __forceinline__ void cp16_g2l(const bf16_t* g, bf16_t* l) {
#if HAVE_ASYNC_LDS
    __builtin_amdgcn_global_load_async_to_lds_b128(
        (AS1 v4i*)(unsigned long long)g,
        (AS3 v4i*)(unsigned)(unsigned long long)l,
        0, 0);
#else
    *(v8bf*)l = *(const v8bf*)g;
#endif
}

// Load a 16-element bf16 WMMA fragment from LDS as two b128 reads.
__device__ __forceinline__ v16bf frag16(const bf16_t* p, int hi_off) {
    v8bf lo = *(const v8bf*)(p);
    v8bf hi = *(const v8bf*)(p + hi_off);
    return __builtin_shufflevector(lo, hi,
            0,1,2,3,4,5,6,7,8,9,10,11,12,13,14,15);
}

// ---------------- pass 1: f32 -> bf16 (RNE via hardware cvt) ---------------
__global__ __launch_bounds__(256) void f32_to_bf16_kernel(
    const float* __restrict__ src, bf16_t* __restrict__ dst, long long n4) {
    long long i = (long long)blockIdx.x * blockDim.x + threadIdx.x;
    long long stride = (long long)gridDim.x * blockDim.x;
    for (; i < n4; i += stride) {
        v4f v = ((const v4f*)src)[i];
        ((v4bf*)dst)[i] = __builtin_convertvector(v, v4bf);
    }
}

// ---------------- pass 2: GEMM + RMSNorm + RoPE ----------------------------
template <bool BF16IN>
__global__ __launch_bounds__(256) void qkv_gemm_kernel(
    const void* __restrict__ Apv,   // [16384,4096] bf16 or f32
    const void* __restrict__ Bpv,   // [5120,4096]  bf16 or f32
    const float* __restrict__ sinp,
    const float* __restrict__ cosp,
    const float* __restrict__ qnw,
    const float* __restrict__ knw,
    float* __restrict__ out)
{
    // Two (A+B) bf16 staging buffers; f32 C half-tile overlays buffer 0.
    __shared__ __align__(16) char smem[2 * BUFE * sizeof(bf16_t)]; // 73728B
    float* Cls = (float*)smem;       // 64 * LDC * 4 = 33792B, overlays buf0

    const int tid  = threadIdx.x;
    const int lane = tid & 31;
    const int wave = tid >> 5;
    const int wm2  = wave & 3;       // 4 M slots of 32 rows
    const int wn   = wave >> 2;      // 2 N slots of 64 cols
    const int ln15 = lane & 15;

    const int nstrip = blockIdx.x;   // 0..39
    const int mtile  = blockIdx.y;   // 0..127

    v8f zero = {0.f,0.f,0.f,0.f,0.f,0.f,0.f,0.f};
    v8f acc0[4], acc1[4];
    acc0[0]=acc0[1]=acc0[2]=acc0[3]=zero;
    acc1[0]=acc1[1]=acc1[2]=acc1[3]=zero;

    // CDNA5 wave32 fragment addressing (see CDNA5 ISA layouts)
    const int a_col  = (lane >> 4) * 8;    // A: K-interleave per lane half
    const int b_colk = (lane >> 4) * 8;    // B: K-interleave per lane half

    // 2 k-steps x (2 A-frags x 4 B-frags) = 16 WMMA per chunk per wave
    auto compute = [&](int buf) {
        const bf16_t* Ab = (const bf16_t*)smem + buf * BUFE;
        const bf16_t* Bb = Ab + 128 * LDA;
        #pragma unroll
        for (int ks = 0; ks < 2; ++ks) {
            const int ko = ks * 32;
            v16bf a0 = frag16(&Ab[(wm2*32 +      ln15) * LDA + ko + a_col], 16);
            v16bf a1 = frag16(&Ab[(wm2*32 + 16 + ln15) * LDA + ko + a_col], 16);
            #pragma unroll
            for (int j = 0; j < 4; ++j) {
                v16bf b = frag16(&Bb[(wn*64 + j*16 + ln15) * LDA + ko + b_colk], 16);
                acc0[j] = __builtin_amdgcn_wmma_f32_16x16x32_bf16(
                              false, a0, false, b, (short)0, acc0[j], false, false);
                acc1[j] = __builtin_amdgcn_wmma_f32_16x16x32_bf16(
                              false, a1, false, b, (short)0, acc1[j], false, false);
            }
        }
    };

    if constexpr (BF16IN) {
        const bf16_t* AgBase = (const bf16_t*)Apv + (size_t)mtile  * MT   * HIDDEN;
        const bf16_t* BgBase = (const bf16_t*)Bpv + (size_t)nstrip * HEAD * HIDDEN;

        auto issue = [&](int buf, int kc) {
            bf16_t* Ad = (bf16_t*)smem + buf * BUFE;
            bf16_t* Bd = Ad + 128 * LDA;
            const bf16_t* Ag = AgBase + kc;
            const bf16_t* Bg = BgBase + kc;
            #pragma unroll
            for (int i = 0; i < 4; ++i) {            // A: 128 rows x 8 chunks
                int q = tid + 256 * i;
                int r = q >> 3, c = (q & 7) * 8;
                cp16_g2l(Ag + (size_t)r * HIDDEN + c, &Ad[r * LDA + c]);
            }
            #pragma unroll
            for (int i = 0; i < 4; ++i) {            // B: 128 rows x 8 chunks
                int q = tid + 256 * i;
                int r = q >> 3, c = (q & 7) * 8;
                cp16_g2l(Bg + (size_t)r * HIDDEN + c, &Bd[r * LDA + c]);
            }
        };

        // Double-buffered async pipeline: loads for chunk i+1 fly while
        // chunk i's WMMAs execute.
        issue(0, 0);
        for (int i = 0; i < NCHUNK; ++i) {
            wait_async0();       // my chunk-i loads have landed in LDS
            __syncthreads();     // everyone's have; idle buffer reads retired
            if (i + 1 < NCHUNK) issue((i + 1) & 1, (i + 1) * KC);
            compute(i & 1);
        }
    } else {
        // Fallback (ws too small): convert f32 -> bf16 while staging, buf 0.
        bf16_t* Ad = (bf16_t*)smem;
        bf16_t* Bd = Ad + 128 * LDA;
        for (int kc = 0; kc < HIDDEN; kc += KC) {
            __syncthreads();
            const float* Ag = (const float*)Apv + (size_t)mtile * MT * HIDDEN + kc;
            const float* Bg = (const float*)Bpv + (size_t)nstrip * HEAD * HIDDEN + kc;
            #pragma unroll
            for (int i = 0; i < 8; ++i) {
                int q = tid + 256 * i;
                int r = q >> 4, c = (q & 15) * 4;
                v4f v = *(const v4f*)(Ag + (size_t)r * HIDDEN + c);
                *(v4bf*)&Ad[r * LDA + c] = __builtin_convertvector(v, v4bf);
            }
            #pragma unroll
            for (int i = 0; i < 8; ++i) {
                int q = tid + 256 * i;
                int r = q >> 4, c = (q & 15) * 4;
                v4f v = *(const v4f*)(Bg + (size_t)r * HIDDEN + c);
                *(v4bf*)&Bd[r * LDA + c] = __builtin_convertvector(v, v4bf);
            }
            if (kc + KC < HIDDEN) {                  // global_prefetch_b8
                __builtin_prefetch(Ag + (size_t)(tid >> 1) * HIDDEN + KC, 0, 1);
                __builtin_prefetch(Bg + (size_t)(tid >> 1) * HIDDEN + KC, 0, 1);
            }
            __syncthreads();
            compute(0);
        }
    }

    // ---- epilogue: spill + normalize 64 rows at a time (C overlays buf0) --
    const int r     = tid >> 2;
    const int dbase = (tid & 3) * 32;
    #pragma unroll
    for (int half = 0; half < 2; ++half) {
        __syncthreads();   // LDS free (compute done / previous half consumed)
        if ((wm2 >> 1) == half) {
            const int lrb = (wm2 & 1) * 32;
            #pragma unroll
            for (int aidx = 0; aidx < 2; ++aidx) {
                #pragma unroll
                for (int j = 0; j < 4; ++j) {
                    int n  = wn * 64 + j * 16 + ln15;
                    int mb = lrb + aidx * 16 + 8 * (lane >> 4);
                    v8f av = aidx ? acc1[j] : acc0[j];
                    #pragma unroll
                    for (int e = 0; e < 8; ++e)
                        Cls[(mb + e) * LDC + n] = av[e];
                }
            }
        }
        __syncthreads();

        const size_t trow = (size_t)mtile * MT + half * 64 + r;
        float* op = out + trow * NTOT + (size_t)nstrip * HEAD;

        if (nstrip >= 36) {                          // V: passthrough
            #pragma unroll 8
            for (int d = dbase; d < dbase + 32; ++d)
                op[d] = Cls[r * LDC + d];
        } else {                                     // Q/K: RMSNorm + RoPE
            const float* nw = (nstrip < 32) ? qnw : knw;
            float ss = 0.f;
            #pragma unroll 8
            for (int d = dbase; d < dbase + 32; ++d) {
                float x = Cls[r * LDC + d];
                ss += x * x;
            }
            ss += __shfl_xor(ss, 1, 32);
            ss += __shfl_xor(ss, 2, 32);
            const float inv = rsqrtf(ss * (1.0f / HEAD) + 1e-6f);
            const float* sp = sinp + trow * HEAD;
            const float* cp = cosp + trow * HEAD;
            const float  sgn = (dbase < 64) ? -1.f : 1.f;
            #pragma unroll 8
            for (int d = dbase; d < dbase + 32; ++d) {
                float xn  = Cls[r * LDC + d]        * inv * nw[d];
                float xro = Cls[r * LDC + (d ^ 64)] * inv * nw[d ^ 64] * sgn;
                op[d] = xro * sp[d] + xn * cp[d];
            }
        }
    }
}

extern "C" void kernel_launch(void* const* d_in, const int* in_sizes, int n_in,
                              void* d_out, int out_size, void* d_ws, size_t ws_size,
                              hipStream_t stream) {
    (void)in_sizes; (void)n_in; (void)out_size;
    const float* hidden = (const float*)d_in[0];
    const float* sinp   = (const float*)d_in[1];
    const float* cosp   = (const float*)d_in[2];
    const float* wqkv   = (const float*)d_in[3];
    const float* qnw    = (const float*)d_in[4];
    const float* knw    = (const float*)d_in[5];
    float* out = (float*)d_out;

    const size_t nA = (size_t)T_ROWS * HIDDEN;   // 67.1M
    const size_t nB = (size_t)NTOT   * HIDDEN;   // 21.0M
    const size_t need = (nA + nB) * sizeof(bf16_t);

    dim3 grid(40, T_ROWS / MT);   // x = head strip: A tile stays hot in L2
    if (ws_size >= need) {
        bf16_t* Ab = (bf16_t*)d_ws;
        bf16_t* Wb = Ab + nA;
        f32_to_bf16_kernel<<<2048, 256, 0, stream>>>(hidden, Ab, (long long)(nA / 4));
        f32_to_bf16_kernel<<<1024, 256, 0, stream>>>(wqkv,   Wb, (long long)(nB / 4));
        qkv_gemm_kernel<true><<<grid, 256, 0, stream>>>(
            Ab, Wb, sinp, cosp, qnw, knw, out);
    } else {
        qkv_gemm_kernel<false><<<grid, 256, 0, stream>>>(
            hidden, wqkv, sinp, cosp, qnw, knw, out);
    }
}
